// GNN_First_Layer_27058293965314
// MI455X (gfx1250) — compile-verified, matrix operand order checked by reading, not run on hardware
//
#include <hip/hip_runtime.h>

#define NATOMS 100000
#define ADIM   12
#define RDIM   1024
#define FDIM   128
#define KNEI   10
#define KEXT   64              // padded ext K: [atoms(12) | mean_same(12) | mean_diff(12) | 0-pad]
#define KTOT   (RDIM + KEXT)   // 1088
#define KTILES (KTOT / 32)     // 34 WMMA K-steps total
#define KCHUNK 544             // K per LDS chunk (17 WMMA steps)
#define CTILES (KCHUNK / 32)   // 17
#define MROWS  32              // output rows per block (2 row-tiles per wave)

typedef __attribute__((ext_vector_type(16))) __bf16        v16bf;
typedef __attribute__((ext_vector_type(8)))  float         v8f;
typedef __attribute__((ext_vector_type(4)))  unsigned int  u32x4;
typedef __attribute__((ext_vector_type(4)))  float         f32x4;

static __device__ __forceinline__ unsigned short f2bf(float f) {
    // round-to-nearest-even fp32 -> bf16
    unsigned int u = __float_as_uint(f);
    u += 0x7fffu + ((u >> 16) & 1u);
    return (unsigned short)(u >> 16);
}

union BFragU {
    struct { u32x4 lo, hi; } u;   // two 16B chunks (the bf16 fragment's 8+8 K-values)
    v16bf v;
};

// ---------------------------------------------------------------------------
// Kernel 1: shuffle [Wr ; Wv ; Wsr ; Wdr ; zeros] (fp32) into bf16 B-fragments
// laid out so each lane's 16 values for one WMMA step are contiguous (32B).
// Layout: Bfrag[(((t*8 + coltile)*32 + lane) * 16) + idx]
// B-matrix bf16 32x16 layout: lanes 0-15 -> col=lane, K = {0..7, 16..23};
//                             lanes 16-31 -> col=lane-16, K = {8..15, 24..31}.
// ---------------------------------------------------------------------------
__global__ void build_bfrag(const float* __restrict__ Wv,  const float* __restrict__ Wr,
                            const float* __restrict__ Wsr, const float* __restrict__ Wdr,
                            unsigned short* __restrict__ Bfrag) {
    const int t  = blockIdx.x;          // K tile 0..33
    const int c  = threadIdx.x >> 5;    // column tile 0..7
    const int L  = threadIdx.x & 31;    // lane
    const int n  = L & 15;
    const int hi = L >> 4;
    const int col = c * 16 + n;
    unsigned short* dst = Bfrag + ((size_t)(t * 8 + c) * 32 + L) * 16;
    #pragma unroll
    for (int idx = 0; idx < 16; ++idx) {
        const int kk = (idx < 8 ? idx : idx + 8) + hi * 8;
        const int k  = t * 32 + kk;
        float v = 0.0f;
        if      (k < RDIM)      v = Wr [(size_t)k * FDIM + col];
        else if (k < RDIM + 12) v = Wv [(size_t)(k - RDIM)      * FDIM + col];
        else if (k < RDIM + 24) v = Wsr[(size_t)(k - RDIM - 12) * FDIM + col];
        else if (k < RDIM + 36) v = Wdr[(size_t)(k - RDIM - 24) * FDIM + col];
        dst[idx] = f2bf(v);
    }
}

// ---------------------------------------------------------------------------
// Kernel 2: per-atom ext row (bf16): [atoms | masked-mean same | masked-mean diff | 0]
// Uses mean_k(atoms[idx] @ W) == (mean_k atoms[idx]) @ W to collapse the
// gather-GEMM into a 12-dim gather-mean feeding the fused GEMM's extra K rows.
// ---------------------------------------------------------------------------
__global__ void build_aext(const float* __restrict__ atoms,
                           const int*   __restrict__ same,
                           const int*   __restrict__ diff,
                           unsigned short* __restrict__ Aext) {
    const int i = blockIdx.x * blockDim.x + threadIdx.x;
    if (i >= NATOMS) return;
    float self[ADIM], ss[ADIM], dd[ADIM];
    #pragma unroll
    for (int a = 0; a < ADIM; ++a) {
        self[a] = atoms[(size_t)i * ADIM + a];
        ss[a] = 0.0f; dd[a] = 0.0f;
    }
    int scnt = 0, dcnt = 0;
    #pragma unroll
    for (int k = 0; k < KNEI; ++k) {
        const int js = same[(size_t)i * KNEI + k];
        if (js > -1) {
            ++scnt;
            #pragma unroll
            for (int a = 0; a < ADIM; ++a) ss[a] += atoms[(size_t)js * ADIM + a];
        }
        const int jd = diff[(size_t)i * KNEI + k];
        if (jd > -1) {
            ++dcnt;
            #pragma unroll
            for (int a = 0; a < ADIM; ++a) dd[a] += atoms[(size_t)jd * ADIM + a];
        }
    }
    const float sn = 1.0f / (float)(scnt > 0 ? scnt : 1);
    const float dn = 1.0f / (float)(dcnt > 0 ? dcnt : 1);
    unsigned short* dst = Aext + (size_t)i * KEXT;
    #pragma unroll
    for (int a = 0; a < ADIM; ++a) {
        dst[a]      = f2bf(self[a]);
        dst[12 + a] = f2bf(ss[a] * sn);
        dst[24 + a] = f2bf(dd[a] * dn);
    }
    #pragma unroll
    for (int z = 36; z < KEXT; ++z) dst[z] = 0;
}

// ---------------------------------------------------------------------------
// Kernel 3: fused GEMM + relu.  out[N,128] = relu(A_ext[N,1088] @ W_cat[1088,128])
// Block = 256 threads = 8 wave32s; block owns MROWS=32 rows, wave w -> 16-col
// tile w with TWO row-tile accumulators, so each B fragment (L2) feeds 2 WMMAs.
// K is processed in two 544-wide chunks so the LDS A-tile stays at 34.8 KB.
// ---------------------------------------------------------------------------
__global__ void __launch_bounds__(256)
gemm_fused(const float* __restrict__ residues,
           const unsigned short* __restrict__ Aext,
           const unsigned short* __restrict__ Bfrag,
           float* __restrict__ out) {
    __shared__ unsigned short lds[MROWS * KCHUNK];   // 32 x 544 bf16 = 34816 B
    const int tid  = threadIdx.x;
    const int row0 = blockIdx.x * MROWS;

    const int wv   = tid >> 5;                  // wave -> column tile
    const int lane = tid & 31;
    const int m    = lane & 15;
    const int hi   = lane >> 4;
    // bf16 A 16x32 layout: lanes 0-15 row=lane K={0..7,16..23}; lanes 16-31 K={8..15,24..31}
    const unsigned short* abase = &lds[m * KCHUNK + hi * 8];
    const u32x4* bbase = (const u32x4*)Bfrag;

    v8f acc0 = {};     // rows row0 +  0..15
    v8f acc1 = {};     // rows row0 + 16..31

    for (int c = 0; c < 2; ++c) {
        if (c) __syncthreads();                 // drain readers before restage
        // --- stage A chunk: 32 rows x 544 bf16 (17408 elems = 4352 quads) ---
        const int kbase = c * KCHUNK;
        #pragma unroll
        for (int it = 0; it < 17; ++it) {
            const int q  = tid + it * 256;      // quad index 0..4351
            const int e  = q * 4;
            const int r  = e / KCHUNK;          // row in tile
            const int kl = e - r * KCHUNK;      // K within chunk (multiple of 4)
            const int kg = kbase + kl;          // global K
            unsigned long long pack;
            if (kg < RDIM) {                    // residues: fp32 -> bf16
                const f32x4 f = *(const f32x4*)(residues + (size_t)(row0 + r) * RDIM + kg);
                pack =  (unsigned long long)f2bf(f[0])        |
                       ((unsigned long long)f2bf(f[1]) << 16) |
                       ((unsigned long long)f2bf(f[2]) << 32) |
                       ((unsigned long long)f2bf(f[3]) << 48);
            } else {                            // ext columns: already bf16
                pack = *(const unsigned long long*)
                           (Aext + (size_t)(row0 + r) * KEXT + (kg - RDIM));
            }
            *(unsigned long long*)&lds[r * KCHUNK + kl] = pack;
        }
        __syncthreads();

        // --- 17 WMMA K-steps; one B fragment feeds both row-tiles ---
        #pragma unroll 2
        for (int t = 0; t < CTILES; ++t) {
            BFragU a0, a1, b;
            a0.u.lo = *(const u32x4*)(abase + t * 32);                      // ds_load_b128
            a0.u.hi = *(const u32x4*)(abase + t * 32 + 16);
            a1.u.lo = *(const u32x4*)(abase + 16 * KCHUNK + t * 32);
            a1.u.hi = *(const u32x4*)(abase + 16 * KCHUNK + t * 32 + 16);
            const int tg = c * CTILES + t;                                  // global K tile
            const size_t bi = ((size_t)(tg * 8 + wv) * 32 + lane) * 2;
            b.u.lo = bbase[bi];                                             // global_load_b128
            b.u.hi = bbase[bi + 1];
            acc0 = __builtin_amdgcn_wmma_f32_16x16x32_bf16(
                       false, a0.v, false, b.v, (short)0, acc0, false, false);
            acc1 = __builtin_amdgcn_wmma_f32_16x16x32_bf16(
                       false, a1.v, false, b.v, (short)0, acc1, false, false);
        }
    }

    // --- relu epilogue; C layout: VGPR v -> M = v + 8*hi, col = lane&15 ---
    #pragma unroll
    for (int v = 0; v < 8; ++v) {
        float x0 = acc0[v];
        float x1 = acc1[v];
        x0 = x0 > 0.0f ? x0 : 0.0f;
        x1 = x1 > 0.0f ? x1 : 0.0f;
        const int r = v + hi * 8;
        out[(size_t)(row0 + r)      * FDIM + wv * 16 + m] = x0;
        out[(size_t)(row0 + 16 + r) * FDIM + wv * 16 + m] = x1;
    }
}

// ---------------------------------------------------------------------------
// Output layout (float slots): res0[N*F] | same0[N*K] | diff0[N*K] | res1[N*F]
//                              | same1[N*K] | diff1[N*K]
// ---------------------------------------------------------------------------
extern "C" void kernel_launch(void* const* d_in, const int* in_sizes, int n_in,
                              void* d_out, int out_size, void* d_ws, size_t ws_size,
                              hipStream_t stream) {
    const float* atoms0    = (const float*)d_in[0];
    const float* residues0 = (const float*)d_in[1];
    const int*   same0     = (const int*)  d_in[2];
    const int*   diff0     = (const int*)  d_in[3];
    const float* atoms1    = (const float*)d_in[4];
    const float* residues1 = (const float*)d_in[5];
    const int*   same1     = (const int*)  d_in[6];
    const int*   diff1     = (const int*)  d_in[7];
    const float* Wv        = (const float*)d_in[8];
    const float* Wr        = (const float*)d_in[9];
    const float* Wsr       = (const float*)d_in[10];
    const float* Wdr       = (const float*)d_in[11];

    // workspace: Bfrag (272 KB, padded to 512 KB) then two bf16 ext buffers
    unsigned short* Bfrag = (unsigned short*)d_ws;
    unsigned short* Aext0 = (unsigned short*)((char*)d_ws + (512u << 10));
    unsigned short* Aext1 = (unsigned short*)((char*)d_ws + (512u << 10) +
                                              (size_t)NATOMS * KEXT * 2);

    float* out = (float*)d_out;
    const size_t NF = (size_t)NATOMS * FDIM;
    const size_t NK = (size_t)NATOMS * KNEI;
    float* res0 = out;
    float* res1 = out + NF + 2 * NK;

    build_bfrag<<<KTILES, 256, 0, stream>>>(Wv, Wr, Wsr, Wdr, Bfrag);

    const int gA = (NATOMS + 255) / 256;
    build_aext<<<gA, 256, 0, stream>>>(atoms0, same0, diff0, Aext0);
    build_aext<<<gA, 256, 0, stream>>>(atoms1, same1, diff1, Aext1);

    const int gG = NATOMS / MROWS;   // 3125, exact
    gemm_fused<<<gG, 256, 0, stream>>>(residues0, Aext0, Bfrag, res0);
    gemm_fused<<<gG, 256, 0, stream>>>(residues1, Aext1, Bfrag, res1);

    // index pass-through outputs (raw device-to-device copies)
    hipMemcpyAsync(out + NF,               same0, NK * sizeof(int), hipMemcpyDeviceToDevice, stream);
    hipMemcpyAsync(out + NF + NK,          diff0, NK * sizeof(int), hipMemcpyDeviceToDevice, stream);
    hipMemcpyAsync(out + 2 * NF + 2 * NK,  same1, NK * sizeof(int), hipMemcpyDeviceToDevice, stream);
    hipMemcpyAsync(out + 2 * NF + 3 * NK,  diff1, NK * sizeof(int), hipMemcpyDeviceToDevice, stream);
}
